// LiteGTNet_65420941853362
// MI455X (gfx1250) — matrix operand order repaired and checked by reading, not compiled, not run on hardware
//
#include <hip/hip_runtime.h>
#include <hip/hip_bf16.h>

// ---------------------------------------------------------------------------
// LiteGT graph-transformer forward for MI455X (gfx1250, wave32, WMMA).
// Dense GEMMs: v_wmma_f32_16x16x32_bf16, K-loop fully unrolled via template,
// 8 column tiles per wave for D>=128 outputs. For the dominant (K=128, CT=8)
// shape the per-block weight panel is staged once into LDS (36KB of the
// 320KB WGP LDS, 32B-aligned + bank-skewed) so the 8 waves read B fragments
// via ds_load_b128 instead of 8x redundant L2 traffic. Fused
// bias/residual/ReLU epilogues. Irregular graph ops use fp32 atomics.
// ---------------------------------------------------------------------------

typedef __attribute__((ext_vector_type(16))) __bf16 v16bf;
typedef __attribute__((ext_vector_type(8)))  float  v8f;

#define NNODES 20000
#define NEDGES 320000
#define DIM    128

static __device__ __forceinline__ __bf16 f2bf(float x) {
  return static_cast<__bf16>(x);
}

// ---------------------------------------------------------------------------
// Weight convert+transpose: W[K,Nc] fp32 row-major  ->  Bt[Ncp, Kld] bf16,
// zero-padded so GEMM B-fragment loads need no bounds checks.
// ---------------------------------------------------------------------------
__global__ void convert_wt(const float* __restrict__ W, __bf16* __restrict__ Bt,
                           int K, int Nc, int Kld, int Ncp) {
  size_t n = (size_t)Ncp * Kld;
  for (size_t i = blockIdx.x * (size_t)blockDim.x + threadIdx.x; i < n;
       i += (size_t)gridDim.x * blockDim.x) {
    int c = (int)(i / Kld);
    int k = (int)(i % Kld);
    float v = (c < Nc && k < K) ? W[(size_t)k * Nc + c] : 0.0f;
    Bt[i] = f2bf(v);
  }
}

__global__ void zero_f32(float* __restrict__ p, size_t n) {
  for (size_t i = blockIdx.x * (size_t)blockDim.x + threadIdx.x; i < n;
       i += (size_t)gridDim.x * blockDim.x)
    p[i] = 0.0f;
}

// ---------------------------------------------------------------------------
// WMMA GEMM:  C[M,Nc] = A[M,K](fp32) * B[K,Nc]  (+bias)(+residual)(relu?)
// B supplied transposed & padded as Bt[Ncp, Kld] bf16.
// Block = 256 threads = 8 waves; wave w -> rows [blk*128 + 16w, +16),
// CT column tiles of 16 per wave (grid.y covers Nc in steps of 16*CT).
// KT > 0: compile-time K (multiple of 32) -> fully unrolled, no K guards.
// KT == 0: runtime K with per-element guards (tiny embedding GEMMs only).
// KT==128 && CT==8: B panel staged in LDS (stride 144 bf16 -> 32B aligned).
// ---------------------------------------------------------------------------
template <int KT, int CT>
__global__ __launch_bounds__(256)
void gemm_t(const float* __restrict__ A, int lda,
            const __bf16* __restrict__ Bt, int ldb,
            float* __restrict__ C, int ldc,
            int M, int Nc, int Krt,
            const float* __restrict__ bias,
            const float* __restrict__ residual, int ldr,
            int relu) {
  constexpr bool USE_LDS = (KT == 128 && CT == 8);
  constexpr int SSTRIDE = 144;  // 128 K + 16 pad (32B-aligned rows, bank skew)
  __shared__ __bf16 sB[USE_LDS ? 128 * SSTRIDE : 1];

  const int K = (KT > 0) ? KT : Krt;
  const int wave = threadIdx.x >> 5;
  const int lane = threadIdx.x & 31;
  const int hf   = lane >> 4;     // which 16-lane half
  const int lrow = lane & 15;
  const int rowBase = blockIdx.x * 128 + wave * 16;
  const int colBase = blockIdx.y * (16 * CT);
  const int r = rowBase + lrow;
  const bool inRow = (r < M);
  const float* arow = A + (size_t)r * lda;

  if constexpr (USE_LDS) {
    // Cooperative 32KB panel load: 128 cols x 128 K bf16, 16B chunks.
    const __bf16* gB = Bt + (size_t)colBase * ldb;
    for (int idx = threadIdx.x; idx < 128 * 16; idx += 256) {
      const int c  = idx >> 4;
      const int kk = (idx & 15) * 8;
      *(uint4*)&sB[c * SSTRIDE + kk] = *(const uint4*)&gB[(size_t)c * ldb + kk];
    }
    __syncthreads();
  }

  v8f zero8 = {};
  v8f acc[CT];
#pragma unroll
  for (int ct = 0; ct < CT; ++ct) acc[ct] = zero8;

#pragma unroll
  for (int k0 = 0; k0 < ((KT > 0) ? KT : 1) * ((KT > 0) ? 1 : 0x40000000);
       k0 += 32) {
    if (KT == 0 && k0 >= K) break;
    const int kb1 = k0 + hf * 16;       // A elements 0..7  : K = kb1+0..7
    const int kb2 = k0 + hf * 16 + 8;   // A elements 8..15 : K = kb2+0..7
    v16bf a;
    if (KT > 0) {
      if (inRow) {
        const float4 f0 = *(const float4*)(arow + kb1);
        const float4 f1 = *(const float4*)(arow + kb1 + 4);
        const float4 f2 = *(const float4*)(arow + kb2);
        const float4 f3 = *(const float4*)(arow + kb2 + 4);
        a[0] = f2bf(f0.x); a[1] = f2bf(f0.y); a[2] = f2bf(f0.z); a[3] = f2bf(f0.w);
        a[4] = f2bf(f1.x); a[5] = f2bf(f1.y); a[6] = f2bf(f1.z); a[7] = f2bf(f1.w);
        a[8]  = f2bf(f2.x); a[9]  = f2bf(f2.y); a[10] = f2bf(f2.z); a[11] = f2bf(f2.w);
        a[12] = f2bf(f3.x); a[13] = f2bf(f3.y); a[14] = f2bf(f3.z); a[15] = f2bf(f3.w);
        if (k0 + 64 < K) __builtin_prefetch(arow + k0 + 64, 0, 3);
      } else {
#pragma unroll
        for (int i = 0; i < 16; ++i) a[i] = f2bf(0.0f);
      }
    } else {
      if (inRow && (k0 + 32 <= K)) {
        const float4 f0 = *(const float4*)(arow + kb1);
        const float4 f1 = *(const float4*)(arow + kb1 + 4);
        const float4 f2 = *(const float4*)(arow + kb2);
        const float4 f3 = *(const float4*)(arow + kb2 + 4);
        a[0] = f2bf(f0.x); a[1] = f2bf(f0.y); a[2] = f2bf(f0.z); a[3] = f2bf(f0.w);
        a[4] = f2bf(f1.x); a[5] = f2bf(f1.y); a[6] = f2bf(f1.z); a[7] = f2bf(f1.w);
        a[8]  = f2bf(f2.x); a[9]  = f2bf(f2.y); a[10] = f2bf(f2.z); a[11] = f2bf(f2.w);
        a[12] = f2bf(f3.x); a[13] = f2bf(f3.y); a[14] = f2bf(f3.z); a[15] = f2bf(f3.w);
      } else {
#pragma unroll
        for (int i = 0; i < 8; ++i) {
          const int k1 = kb1 + i, k2 = kb2 + i;
          a[i]     = (inRow && k1 < K) ? f2bf(arow[k1]) : f2bf(0.0f);
          a[8 + i] = (inRow && k2 < K) ? f2bf(arow[k2]) : f2bf(0.0f);
        }
      }
    }
#pragma unroll
    for (int ct = 0; ct < CT; ++ct) {
      v16bf b;
      if constexpr (USE_LDS) {
        const int cl = ct * 16 + lrow;  // local column within panel
        b = *(const v16bf*)&sB[cl * SSTRIDE + k0 + hf * 16];  // ds_load_b128 x2
      } else {
        const int c = colBase + ct * 16 + lrow;  // padded: in-bounds
        b = *(const v16bf*)(Bt + (size_t)c * ldb + k0 + hf * 16);
      }
      acc[ct] = __builtin_amdgcn_wmma_f32_16x16x32_bf16(
          false, a, false, b, (short)0, acc[ct], false, false);
    }
  }

  // Epilogue: accumulator VGPR v holds row = rowBase + v + hf*8, col c.
#pragma unroll
  for (int ct = 0; ct < CT; ++ct) {
    const int c = colBase + ct * 16 + lrow;
    if (c >= Nc) continue;
    const float bv = bias ? bias[c] : 0.0f;
#pragma unroll
    for (int v = 0; v < 8; ++v) {
      const int row = rowBase + hf * 8 + v;
      if (row >= M) continue;
      float val = acc[ct][v] + bv;
      if (residual) val += residual[(size_t)row * ldr + c];
      if (relu) val = fmaxf(val, 0.0f);
      C[(size_t)row * ldc + c] = val;
    }
  }
}

// ---------------------------------------------------------------------------
// Graph / elementwise kernels
// ---------------------------------------------------------------------------
__global__ void deg_count(const int* __restrict__ dst, float* __restrict__ deg, int E) {
  for (int e = blockIdx.x * blockDim.x + threadIdx.x; e < E;
       e += gridDim.x * blockDim.x)
    atomicAdd(&deg[dst[e]], 1.0f);
}

__global__ void score_d_k(const int* __restrict__ dst, const float* __restrict__ deg,
                          float* __restrict__ sd, int E) {
  for (int e = blockIdx.x * blockDim.x + threadIdx.x; e < E;
       e += gridDim.x * blockDim.x)
    sd[e] = 1.0f / (fmaxf(deg[dst[e]], 1.0f) * 16.0f);
}

// One wave per edge. 4 dims/lane; head h = lane/4 covers dims 16h..16h+15.
__global__ __launch_bounds__(256)
void attn_edge(const float* __restrict__ Q, const float* __restrict__ Kf,
               const float* __restrict__ V, float* __restrict__ Ep,
               const int* __restrict__ src, const int* __restrict__ dst,
               const float* __restrict__ scoreD,
               float* __restrict__ wV, float* __restrict__ z, int E) {
  const int wave = threadIdx.x >> 5;
  const int lane = threadIdx.x & 31;
  const int e = blockIdx.x * 8 + wave;
  if (e >= E) return;
  const int s = src[e], d = dst[e];
  const float scale = 0.25f;  // 1/sqrt(DH), DH=16
  const int o = lane * 4;
  const float4 q  = *(const float4*)(Q  + (size_t)d * DIM + o);
  const float4 k  = *(const float4*)(Kf + (size_t)s * DIM + o);
  const float4 ep = *(const float4*)(Ep + (size_t)e * DIM + o);
  float4 sc;
  sc.x = q.x * k.x * scale * ep.x;
  sc.y = q.y * k.y * scale * ep.y;
  sc.z = q.z * k.z * scale * ep.z;
  sc.w = q.w * k.w * scale * ep.w;
  *(float4*)(Ep + (size_t)e * DIM + o) = sc;  // e_att (score) in place
  float ssum = sc.x + sc.y + sc.z + sc.w;
  ssum += __shfl_xor(ssum, 1, 32);            // reduce 4-lane head group
  ssum += __shfl_xor(ssum, 2, 32);
  const float a = __expf(fminf(fmaxf(ssum, -5.0f), 5.0f)) + scoreD[e];
  const float4 v = *(const float4*)(V + (size_t)s * DIM + o);
  atomicAdd(&wV[(size_t)d * DIM + o + 0], a * v.x);
  atomicAdd(&wV[(size_t)d * DIM + o + 1], a * v.y);
  atomicAdd(&wV[(size_t)d * DIM + o + 2], a * v.z);
  atomicAdd(&wV[(size_t)d * DIM + o + 3], a * v.w);
  if ((lane & 3) == 0) atomicAdd(&z[(size_t)d * 8 + (lane >> 2)], a);
}

__global__ void hatt_div(float* __restrict__ wV, const float* __restrict__ z, size_t n) {
  for (size_t i = blockIdx.x * (size_t)blockDim.x + threadIdx.x; i < n;
       i += (size_t)gridDim.x * blockDim.x) {
    const size_t node = i >> 7;
    const int head = (int)((i & 127) >> 4);
    wV[i] = wV[i] / (z[node * 8 + head] + 1e-6f);
  }
}

// BatchNorm over rows: stats[c] = sum, stats[128+c] = sumsq
__global__ __launch_bounds__(128)
void bn_reduce(const float* __restrict__ X, int M, float* __restrict__ stats) {
  const int c = threadIdx.x;  // blockDim = 128
  float s = 0.0f, sq = 0.0f;
  for (int r = blockIdx.x; r < M; r += gridDim.x) {
    const float v = X[(size_t)r * DIM + c];
    s += v; sq += v * v;
  }
  atomicAdd(&stats[c], s);
  atomicAdd(&stats[DIM + c], sq);
}

__global__ void bn_norm(const float* __restrict__ X, float* __restrict__ Y,
                        size_t n, const float* __restrict__ stats, float invM) {
  for (size_t i = blockIdx.x * (size_t)blockDim.x + threadIdx.x; i < n;
       i += (size_t)gridDim.x * blockDim.x) {
    const int c = (int)(i & 127);
    const float m  = stats[c] * invM;
    const float vv = stats[DIM + c] * invM - m * m;
    Y[i] = (X[i] - m) * rsqrtf(vv + 1e-5f);
  }
}

// feat[e] = concat(h[src[e]], h[dst[e]]) ; processed as float4 (64 per row)
__global__ void feat_gather(const float* __restrict__ h, const int* __restrict__ src,
                            const int* __restrict__ dst, float* __restrict__ feat, int E) {
  const float4* h4 = (const float4*)h;
  float4* f4 = (float4*)feat;
  const size_t n = (size_t)E * 64;
  for (size_t i = blockIdx.x * (size_t)blockDim.x + threadIdx.x; i < n;
       i += (size_t)gridDim.x * blockDim.x) {
    const int e = (int)(i >> 6);
    const int j = (int)(i & 63);
    f4[i] = (j < 32) ? h4[(size_t)src[e] * 32 + j]
                     : h4[(size_t)dst[e] * 32 + (j - 32)];
  }
}

// ---------------------------------------------------------------------------
// Host orchestration
// ---------------------------------------------------------------------------
extern "C" void kernel_launch(void* const* d_in, const int* in_sizes, int n_in,
                              void* d_out, int out_size, void* d_ws, size_t ws_size,
                              hipStream_t stream) {
  (void)in_sizes; (void)n_in; (void)out_size; (void)ws_size;
  const int N = NNODES, E = NEDGES, L = 4;

  const float* h_in = (const float*)d_in[0];
  const float* e_in = (const float*)d_in[1];
  const float* lap  = (const float*)d_in[2];
  const int*   src  = (const int*)d_in[3];
  const int*   dst  = (const int*)d_in[4];
  const float* Wh   = (const float*)d_in[5];
  const float* bh   = (const float*)d_in[6];
  const float* Wlap = (const float*)d_in[7];
  const float* We0  = (const float*)d_in[8];
  const float* be0  = (const float*)d_in[9];
  const float* Wq   = (const float*)d_in[10];
  const float* Wk   = (const float*)d_in[11];
  const float* Wv   = (const float*)d_in[12];
  const float* Wep  = (const float*)d_in[13];
  const float* Wo   = (const float*)d_in[14];
  const float* Woe  = (const float*)d_in[15];
  const float* W1   = (const float*)d_in[16];
  const float* W2   = (const float*)d_in[17];
  const float* We1  = (const float*)d_in[18];
  const float* We2  = (const float*)d_in[19];
  const float* M0   = (const float*)d_in[20];
  const float* b0   = (const float*)d_in[21];
  const float* M1   = (const float*)d_in[22];
  const float* b1   = (const float*)d_in[23];
  const float* M2   = (const float*)d_in[24];
  const float* b2   = (const float*)d_in[25];
  float* out = (float*)d_out;

  // bump allocator over workspace (256B aligned)
  char* ws = (char*)d_ws;
  size_t off = 0;
  auto alloc = [&](size_t bytes) -> void* {
    void* p = ws + off;
    off += (bytes + 255) & ~(size_t)255;
    return p;
  };
  auto allocF = [&](size_t n) -> float* { return (float*)alloc(n * sizeof(float)); };
  auto padK = [](int K) { return (K + 31) & ~31; };
  auto padN = [](int Nc) { return (Nc % 128 == 0) ? Nc : ((Nc + 63) & ~63); };
  auto allocW = [&](int K, int Nc) -> __bf16* {
    return (__bf16*)alloc((size_t)padK(K) * padN(Nc) * sizeof(__bf16));
  };

  float* hbuf  = allocF((size_t)N * 128);
  float* ebuf  = allocF((size_t)E * 128);
  float* Qb    = allocF((size_t)N * 128);
  float* Kb    = allocF((size_t)N * 128);
  float* Vb    = allocF((size_t)N * 128);
  float* Epb   = allocF((size_t)E * 128);   // proj_e -> e_att ; later x1
  float* wVb   = allocF((size_t)N * 128);
  float* zb    = allocF((size_t)N * 8);
  float* ffnh  = allocF((size_t)N * 256);
  float* tmph  = allocF((size_t)N * 128);
  float* te128 = allocF((size_t)E * 128);   // edge BN input ; later x2
  float* ffne  = allocF((size_t)E * 256);   // edge FFN hidden ; later feat
  float* deg   = allocF((size_t)N);
  float* sD    = allocF((size_t)E);
  float* stats = allocF(256);

  __bf16* WhT   = allocW(128, 128);
  __bf16* WlapT = allocW(8, 128);
  __bf16* We0T  = allocW(2, 128);
  __bf16 *WqT[4], *WkT[4], *WvT[4], *WepT[4], *WoT[4], *WoeT[4];
  __bf16 *W1T[4], *W2T[4], *We1T[4], *We2T[4];
  for (int l = 0; l < L; ++l) {
    WqT[l]  = allocW(128, 128);  WkT[l]  = allocW(128, 128);
    WvT[l]  = allocW(128, 128);  WepT[l] = allocW(128, 128);
    WoT[l]  = allocW(128, 128);  WoeT[l] = allocW(128, 128);
    W1T[l]  = allocW(128, 256);  W2T[l]  = allocW(256, 128);
    We1T[l] = allocW(128, 256);  We2T[l] = allocW(256, 128);
  }
  __bf16* M0T = allocW(256, 128);
  __bf16* M1T = allocW(128, 64);
  __bf16* M2T = allocW(64, 2);

  auto conv = [&](const float* W, __bf16* Bt, int K, int Nc) {
    const size_t n = (size_t)padK(K) * padN(Nc);
    convert_wt<<<(unsigned)((n + 255) / 256), 256, 0, stream>>>(
        W, Bt, K, Nc, padK(K), padN(Nc));
  };
  auto gemm = [&](const float* A, int lda, const __bf16* Bt, int K,
                  float* C, int M, int Nc, const float* bias,
                  const float* res, int ldr, int relu) {
    const int Kld = padK(K);
    const bool big = (Nc % 128 == 0) && (K == 128 || K == 256);
    const int tiles = big ? 8 : 4;
    dim3 g((M + 127) / 128, (Nc + 16 * tiles - 1) / (16 * tiles));
    if (big && K == 128)
      gemm_t<128, 8><<<g, 256, 0, stream>>>(A, lda, Bt, Kld, C, Nc, M, Nc, K,
                                            bias, res, ldr, relu);
    else if (big)
      gemm_t<256, 8><<<g, 256, 0, stream>>>(A, lda, Bt, Kld, C, Nc, M, Nc, K,
                                            bias, res, ldr, relu);
    else if (K == 128)
      gemm_t<128, 4><<<g, 256, 0, stream>>>(A, lda, Bt, Kld, C, Nc, M, Nc, K,
                                            bias, res, ldr, relu);
    else if (K == 64)
      gemm_t<64, 4><<<g, 256, 0, stream>>>(A, lda, Bt, Kld, C, Nc, M, Nc, K,
                                           bias, res, ldr, relu);
    else
      gemm_t<0, 4><<<g, 256, 0, stream>>>(A, lda, Bt, Kld, C, Nc, M, Nc, K,
                                          bias, res, ldr, relu);
  };
  auto zero = [&](float* p, size_t n) {
    unsigned b = (unsigned)((n + 255) / 256);
    if (b > 4096) b = 4096;
    zero_f32<<<b, 256, 0, stream>>>(p, n);
  };
  auto bn = [&](const float* X, float* Y, int M) {
    zero(stats, 256);
    bn_reduce<<<512, 128, 0, stream>>>(X, M, stats);
    bn_norm<<<4096, 256, 0, stream>>>(X, Y, (size_t)M * 128, stats, 1.0f / (float)M);
  };

  // ---- weight conversion (L2-resident bf16, transposed+padded) ----
  conv(Wh, WhT, 128, 128);
  conv(Wlap, WlapT, 8, 128);
  conv(We0, We0T, 2, 128);
  for (int l = 0; l < L; ++l) {
    conv(Wq  + (size_t)l * 128 * 128, WqT[l],  128, 128);
    conv(Wk  + (size_t)l * 128 * 128, WkT[l],  128, 128);
    conv(Wv  + (size_t)l * 128 * 128, WvT[l],  128, 128);
    conv(Wep + (size_t)l * 128 * 128, WepT[l], 128, 128);
    conv(Wo  + (size_t)l * 128 * 128, WoT[l],  128, 128);
    conv(Woe + (size_t)l * 128 * 128, WoeT[l], 128, 128);
    conv(W1  + (size_t)l * 128 * 256, W1T[l],  128, 256);
    conv(W2  + (size_t)l * 256 * 128, W2T[l],  256, 128);
    conv(We1 + (size_t)l * 128 * 256, We1T[l], 128, 256);
    conv(We2 + (size_t)l * 256 * 128, We2T[l], 256, 128);
  }
  conv(M0, M0T, 256, 128);
  conv(M1, M1T, 128, 64);
  conv(M2, M2T, 64, 2);

  // ---- input embeddings ----
  gemm(h_in, 128, WhT, 128, hbuf, N, 128, bh, nullptr, 0, 0);
  gemm(lap, 8, WlapT, 8, hbuf, N, 128, nullptr, hbuf, 128, 0);  // += lap@Wlap
  gemm(e_in, 2, We0T, 2, ebuf, E, 128, be0, nullptr, 0, 0);

  // ---- degrees / default score ----
  zero(deg, (size_t)N);
  deg_count<<<1024, 256, 0, stream>>>(dst, deg, E);
  score_d_k<<<1024, 256, 0, stream>>>(dst, deg, sD, E);

  // ---- transformer layers ----
  for (int l = 0; l < L; ++l) {
    gemm(hbuf, 128, WqT[l], 128, Qb, N, 128, nullptr, nullptr, 0, 0);
    gemm(hbuf, 128, WkT[l], 128, Kb, N, 128, nullptr, nullptr, 0, 0);
    gemm(hbuf, 128, WvT[l], 128, Vb, N, 128, nullptr, nullptr, 0, 0);
    gemm(ebuf, 128, WepT[l], 128, Epb, E, 128, nullptr, nullptr, 0, 0);

    zero(wVb, (size_t)N * 128);
    zero(zb, (size_t)N * 8);
    attn_edge<<<(E + 7) / 8, 256, 0, stream>>>(Qb, Kb, Vb, Epb, src, dst, sD,
                                               wVb, zb, E);
    hatt_div<<<2048, 256, 0, stream>>>(wVb, zb, (size_t)N * 128);

    // h = BN(h + h_att @ Wo)
    gemm(wVb, 128, WoT[l], 128, tmph, N, 128, nullptr, hbuf, 128, 0);
    bn(tmph, hbuf, N);
    // h = BN(h + relu(h@W1)@W2)
    gemm(hbuf, 128, W1T[l], 128, ffnh, N, 256, nullptr, nullptr, 0, 1);
    gemm(ffnh, 256, W2T[l], 256, tmph, N, 128, nullptr, hbuf, 128, 0);
    bn(tmph, hbuf, N);
    // e = BN(e + e_att @ Woe)
    gemm(Epb, 128, WoeT[l], 128, te128, E, 128, nullptr, ebuf, 128, 0);
    bn(te128, ebuf, E);
    // e = BN(e + relu(e@We1)@We2)
    gemm(ebuf, 128, We1T[l], 128, ffne, E, 256, nullptr, nullptr, 0, 1);
    gemm(ffne, 256, We2T[l], 256, te128, E, 128, nullptr, ebuf, 128, 0);
    bn(te128, ebuf, E);
  }

  // ---- edge readout MLP ----
  feat_gather<<<4096, 256, 0, stream>>>(hbuf, src, dst, ffne, E);  // [E,256]
  gemm(ffne, 256, M0T, 256, Epb, E, 128, b0, nullptr, 0, 1);        // x1 [E,128]
  gemm(Epb, 128, M1T, 128, te128, E, 64, b1, nullptr, 0, 1);        // x2 [E,64]
  gemm(te128, 64, M2T, 64, out, E, 2, b2, nullptr, 0, 0);           // [E,2]
}
